// GroupedLinear_4758823764460
// MI455X (gfx1250) — matrix operand order repaired and checked by reading, not compiled, not run on hardware
//
#include <hip/hip_runtime.h>

typedef __attribute__((ext_vector_type(16))) __bf16        v16bf;
typedef __attribute__((ext_vector_type(8)))  float         v8f;
typedef __attribute__((ext_vector_type(4)))  float         v4f;
typedef __attribute__((ext_vector_type(4)))  unsigned int  v4u;
typedef __attribute__((ext_vector_type(4)))  int           v4i;

#define G_NUM    8
#define T_TOK    16384
#define IN_DIM   1024
#define OUT_DIM  4096
#define CAP      4096

#define BM       128
#define BN       128

// ---- bf16-source (fast) path: BK=64, double-buffered LDS ----
#define BK2      64
#define NK2      (IN_DIM / BK2)     // 16 K tiles
#define PITCH2   72                 // halfwords/row: 64 data + 8 pad -> conflict-free b128 reads
#define TILE_E   (BM * PITCH2)      // elements per matrix tile buffer (9216)

// ---- f32-source fallback path: BK=32, single-buffered ----
#define BK1      32
#define NK1      (IN_DIM / BK1)
#define PITCH1   40

// ---- CDNA5 async global->LDS availability ----
#if defined(__has_builtin)
# if __has_builtin(__builtin_amdgcn_global_load_async_to_lds_b128)
#  define HAVE_ASYNC_LDS 1
# endif
#endif
#ifndef HAVE_ASYNC_LDS
# define HAVE_ASYNC_LDS 0
#endif

struct Frag { union { v16bf v; v4u u[2]; }; };

// ---------------------------------------------------------------------------
// fp32 -> packed bf16 pair (RNE) -- prefer HW packed convert when available
// ---------------------------------------------------------------------------
__device__ __forceinline__ unsigned int pack_bf16_pair(float lo, float hi) {
#if defined(__has_builtin) && __has_builtin(__builtin_amdgcn_cvt_pk_bf16_f32)
    auto t = __builtin_amdgcn_cvt_pk_bf16_f32(lo, hi);
    unsigned int u; __builtin_memcpy(&u, &t, 4);
    return u;
#else
    union { __bf16 b[2]; unsigned int u; } r;
    r.b[0] = (__bf16)lo; r.b[1] = (__bf16)hi;
    return r.u;
#endif
}

// ===========================================================================
// one-time fp32 -> bf16 conversion pass (memory bound)
// ===========================================================================
__global__ void cvt_f32_to_bf16_kernel(const float* __restrict__ src,
                                       __bf16* __restrict__ dst, size_t n)
{
    size_t stride = (size_t)gridDim.x * blockDim.x * 8;
    for (size_t i = ((size_t)blockIdx.x * blockDim.x + threadIdx.x) * 8; i < n; i += stride) {
        v4f a = *(const v4f*)(src + i);
        v4f b = *(const v4f*)(src + i + 4);
        v4u o;
        o.x = pack_bf16_pair(a.x, a.y);
        o.y = pack_bf16_pair(a.z, a.w);
        o.z = pack_bf16_pair(b.x, b.y);
        o.w = pack_bf16_pair(b.z, b.w);
        *(v4u*)(dst + i) = o;
    }
}

#if HAVE_ASYNC_LDS
// Copy 64 contiguous bytes global->LDS via GLOBAL_LOAD_ASYNC_TO_LDS_B128.
// Builtin signature (from toolchain diagnostic): (AS1 v4i*, AS3 v4i*, Imm, Imm).
// Per ISA 08_async_tensor §4.4, the instruction offset is applied to both the
// global and LDS addresses, so one base pair serves all four b128 transfers.
__device__ __forceinline__ void async_copy_64B(const __bf16* g, const __bf16* l) {
    __attribute__((address_space(1))) v4i* gp =
        (__attribute__((address_space(1))) v4i*)(unsigned long long)(size_t)g;
    __attribute__((address_space(3))) v4i* lp =
        (__attribute__((address_space(3))) v4i*)(unsigned int)(size_t)l;
    __builtin_amdgcn_global_load_async_to_lds_b128(gp, lp, 0, 0);
    __builtin_amdgcn_global_load_async_to_lds_b128(gp, lp, 16, 0);
    __builtin_amdgcn_global_load_async_to_lds_b128(gp, lp, 32, 0);
    __builtin_amdgcn_global_load_async_to_lds_b128(gp, lp, 48, 0);
}
#define WAIT_ASYNC0() asm volatile("s_wait_asynccnt 0" ::: "memory")
#endif

__device__ __forceinline__ void stage_ld_bf16(v4u r[4], const __bf16* p, bool valid) {
    if (valid) {
        const v4u* q = (const v4u*)p;           // 64 contiguous bytes
        r[0] = q[0]; r[1] = q[1]; r[2] = q[2]; r[3] = q[3];
    } else {
        v4u z = {0u, 0u, 0u, 0u};
        r[0] = z; r[1] = z; r[2] = z; r[3] = z;
    }
}

// ===========================================================================
// FAST PATH: grouped GEMM, bf16 operands from workspace, fp32 accumulate
// y[off_g + m, n] = sum_k xb[off_g+m,k] * wb[g,n,k] + bias[n]
// ===========================================================================
__global__ __launch_bounds__(256)
void grouped_gemm_bf16(const __bf16* __restrict__ xb,
                       const __bf16* __restrict__ wb,
                       const float*  __restrict__ bias,
                       const int*    __restrict__ lens,
                       float*        __restrict__ y)
{
    const int nt = blockIdx.x;
    const int mt = blockIdx.y;
    const int g  = blockIdx.z;

    const int len = lens[g];
    if (mt * BM >= len) return;

    int off = 0;
    #pragma unroll
    for (int i = 0; i < G_NUM; ++i) off += (i < g) ? lens[i] : 0;

    extern __shared__ __bf16 smem[];            // 2 bufs * (A tile + B tile)
    __bf16* At = smem;                          // At buf at  buf*TILE_E
    __bf16* Bt = smem + 2 * TILE_E;             // Bt buf at  buf*TILE_E

    const int tid    = threadIdx.x;
    const int ldRow  = tid >> 1;                // 0..127
    const int ldHalf = (tid & 1) * 32;          // bf16 col 0 or 32 within BK2

    const bool a_valid = (mt * BM + ldRow) < len;
    const __bf16* aptr = xb + (size_t)(off + mt * BM + ldRow) * IN_DIM + ldHalf;
    const __bf16* bptr = wb + ((size_t)g * OUT_DIM + (size_t)nt * BN + ldRow) * IN_DIM + ldHalf;

    const int wave  = tid >> 5;
    const int lane  = tid & 31;
    const int wm    = wave >> 2;                // 0..1 : 64 rows
    const int wn    = wave & 3;                 // 0..3 : 32 cols
    const int lhalf = lane >> 4;
    const int l16   = lane & 15;

    v8f acc[4][2];
    #pragma unroll
    for (int i = 0; i < 4; ++i)
        #pragma unroll
        for (int j = 0; j < 2; ++j)
            acc[i][j] = (v8f){0.f,0.f,0.f,0.f,0.f,0.f,0.f,0.f};

    float bv[2];
    #pragma unroll
    for (int j = 0; j < 2; ++j)
        bv[j] = bias[nt * BN + wn * 32 + j * 16 + l16];

#if HAVE_ASYNC_LDS
    __bf16* aLds = &At[ldRow * PITCH2 + ldHalf];   // thread-exclusive 64B region
    __bf16* bLds = &Bt[ldRow * PITCH2 + ldHalf];

    // ragged rows: zero this thread's region in both buffers once; async loads
    // for these lanes are never issued, so the zeros persist.
    if (!a_valid) {
        v4u z = {0u,0u,0u,0u};
        v4u* p0 = (v4u*)aLds;
        v4u* p1 = (v4u*)(aLds + TILE_E);
        p0[0]=z; p0[1]=z; p0[2]=z; p0[3]=z;
        p1[0]=z; p1[1]=z; p1[2]=z; p1[3]=z;
    }
    if (a_valid) async_copy_64B(aptr, aLds);
    async_copy_64B(bptr, bLds);
    WAIT_ASYNC0();
    __syncthreads();
#else
    v4u ar[4], br[4];
    stage_ld_bf16(ar, aptr, a_valid);
    stage_ld_bf16(br, bptr, true);
#endif

    for (int kt = 0; kt < NK2; ++kt) {
        const int buf = (kt & 1) * TILE_E;

#if HAVE_ASYNC_LDS
        // issue next tile's async copies first so they overlap the WMMA work
        if (kt + 1 < NK2) {
            const int nbuf = ((kt + 1) & 1) * TILE_E;
            if (a_valid) async_copy_64B(aptr + (kt + 1) * BK2, aLds + nbuf);
            async_copy_64B(bptr + (kt + 1) * BK2, bLds + nbuf);
        }
#else
        v4u* pa = (v4u*)&At[buf + ldRow * PITCH2 + ldHalf];
        pa[0] = ar[0]; pa[1] = ar[1]; pa[2] = ar[2]; pa[3] = ar[3];
        v4u* pb = (v4u*)&Bt[buf + ldRow * PITCH2 + ldHalf];
        pb[0] = br[0]; pb[1] = br[1]; pb[2] = br[2]; pb[3] = br[3];
        __syncthreads();
        if (kt + 1 < NK2) {
            stage_ld_bf16(ar, aptr + (kt + 1) * BK2, a_valid);
            stage_ld_bf16(br, bptr + (kt + 1) * BK2, true);
        }
#endif

        #pragma unroll
        for (int ks = 0; ks < 2; ++ks) {        // two K=32 WMMA steps per tile
            Frag af[4];
            #pragma unroll
            for (int i = 0; i < 4; ++i) {
                const int r    = wm * 64 + i * 16 + l16;
                const int base = buf + r * PITCH2 + ks * 32 + lhalf * 8;
                af[i].u[0] = *(const v4u*)&At[base];
                af[i].u[1] = *(const v4u*)&At[base + 16];
            }
            Frag bfr[2];
            #pragma unroll
            for (int j = 0; j < 2; ++j) {
                const int r    = wn * 32 + j * 16 + l16;
                const int base = buf + r * PITCH2 + ks * 32 + lhalf * 8;
                bfr[j].u[0] = *(const v4u*)&Bt[base];
                bfr[j].u[1] = *(const v4u*)&Bt[base + 16];
            }
            #pragma unroll
            for (int i = 0; i < 4; ++i)
                #pragma unroll
                for (int j = 0; j < 2; ++j)
                    acc[i][j] = __builtin_amdgcn_wmma_f32_16x16x32_bf16(
                        false, af[i].v, false, bfr[j].v,
                        (short)0, acc[i][j], false, false);
        }

#if HAVE_ASYNC_LDS
        if (kt + 1 < NK2) {
            WAIT_ASYNC0();       // own async writes for tile kt+1 committed
            __syncthreads();     // + everyone done reading buf[kt&1]
        }
#endif
    }

    #pragma unroll
    for (int i = 0; i < 4; ++i) {
        const int mbase = mt * BM + wm * 64 + i * 16 + lhalf * 8;
        #pragma unroll
        for (int j = 0; j < 2; ++j) {
            const int ncol = nt * BN + wn * 32 + j * 16 + l16;
            #pragma unroll
            for (int v = 0; v < 8; ++v) {
                const int m = mbase + v;
                if (m < len)
                    y[(size_t)(off + m) * OUT_DIM + ncol] = acc[i][j][v] + bv[j];
            }
        }
    }
}

// ===========================================================================
// FALLBACK PATH (ws too small): convert fp32->bf16 while staging (BK=32)
// ===========================================================================
__device__ __forceinline__ void ld16f(v4f r[4], const float* __restrict__ p, bool valid) {
    if (valid) {
        r[0] = *(const v4f*)(p + 0);
        r[1] = *(const v4f*)(p + 4);
        r[2] = *(const v4f*)(p + 8);
        r[3] = *(const v4f*)(p + 12);
    } else {
        v4f z = {0.f,0.f,0.f,0.f};
        r[0] = z; r[1] = z; r[2] = z; r[3] = z;
    }
}

__device__ __forceinline__ v4u pack8(const v4f a, const v4f b) {
    v4u r;
    r.x = pack_bf16_pair(a.x, a.y);
    r.y = pack_bf16_pair(a.z, a.w);
    r.z = pack_bf16_pair(b.x, b.y);
    r.w = pack_bf16_pair(b.z, b.w);
    return r;
}

__global__ __launch_bounds__(256)
void grouped_gemm_f32src(const float* __restrict__ x,
                         const float* __restrict__ w,
                         const float* __restrict__ bias,
                         const int*   __restrict__ lens,
                         float*       __restrict__ y)
{
    const int nt = blockIdx.x, mt = blockIdx.y, g = blockIdx.z;
    const int len = lens[g];
    if (mt * BM >= len) return;
    int off = 0;
    #pragma unroll
    for (int i = 0; i < G_NUM; ++i) off += (i < g) ? lens[i] : 0;

    __shared__ unsigned short Atile[BM * PITCH1];
    __shared__ unsigned short Btile[BN * PITCH1];

    const int tid    = threadIdx.x;
    const int ldRow  = tid >> 1;
    const int ldHalf = (tid & 1) * 16;

    const bool a_valid = (mt * BM + ldRow) < len;
    const float* aptr = x + (size_t)(off + mt * BM + ldRow) * IN_DIM + ldHalf;
    const float* bptr = w + ((size_t)g * OUT_DIM + (size_t)nt * BN + ldRow) * IN_DIM + ldHalf;

    const int wave = tid >> 5, lane = tid & 31;
    const int wm = wave >> 2, wn = wave & 3;
    const int lhalf = lane >> 4, l16 = lane & 15;

    v8f acc[4][2];
    #pragma unroll
    for (int i = 0; i < 4; ++i)
        #pragma unroll
        for (int j = 0; j < 2; ++j)
            acc[i][j] = (v8f){0.f,0.f,0.f,0.f,0.f,0.f,0.f,0.f};

    float bv[2];
    #pragma unroll
    for (int j = 0; j < 2; ++j) bv[j] = bias[nt * BN + wn * 32 + j * 16 + l16];

    v4f ar[4], br[4];
    ld16f(ar, aptr, a_valid);
    ld16f(br, bptr, true);

    for (int kt = 0; kt < NK1; ++kt) {
        const v4u a0 = pack8(ar[0], ar[1]), a1 = pack8(ar[2], ar[3]);
        const v4u b0 = pack8(br[0], br[1]), b1 = pack8(br[2], br[3]);
        __syncthreads();
        *(v4u*)&Atile[ldRow * PITCH1 + ldHalf]     = a0;
        *(v4u*)&Atile[ldRow * PITCH1 + ldHalf + 8] = a1;
        *(v4u*)&Btile[ldRow * PITCH1 + ldHalf]     = b0;
        *(v4u*)&Btile[ldRow * PITCH1 + ldHalf + 8] = b1;
        __syncthreads();
        if (kt + 1 < NK1) {
            ld16f(ar, aptr + (kt + 1) * BK1, a_valid);
            ld16f(br, bptr + (kt + 1) * BK1, true);
        }
        Frag af[4];
        #pragma unroll
        for (int i = 0; i < 4; ++i) {
            const int base = (wm * 64 + i * 16 + l16) * PITCH1 + lhalf * 8;
            af[i].u[0] = *(const v4u*)&Atile[base];
            af[i].u[1] = *(const v4u*)&Atile[base + 16];
        }
        Frag bfr[2];
        #pragma unroll
        for (int j = 0; j < 2; ++j) {
            const int base = (wn * 32 + j * 16 + l16) * PITCH1 + lhalf * 8;
            bfr[j].u[0] = *(const v4u*)&Btile[base];
            bfr[j].u[1] = *(const v4u*)&Btile[base + 16];
        }
        #pragma unroll
        for (int i = 0; i < 4; ++i)
            #pragma unroll
            for (int j = 0; j < 2; ++j)
                acc[i][j] = __builtin_amdgcn_wmma_f32_16x16x32_bf16(
                    false, af[i].v, false, bfr[j].v, (short)0, acc[i][j], false, false);
    }

    #pragma unroll
    for (int i = 0; i < 4; ++i) {
        const int mbase = mt * BM + wm * 64 + i * 16 + lhalf * 8;
        #pragma unroll
        for (int j = 0; j < 2; ++j) {
            const int ncol = nt * BN + wn * 32 + j * 16 + l16;
            #pragma unroll
            for (int v = 0; v < 8; ++v) {
                const int m = mbase + v;
                if (m < len)
                    y[(size_t)(off + m) * OUT_DIM + ncol] = acc[i][j][v] + bv[j];
            }
        }
    }
}

// Zero rows beyond sum(group_lens) (d_out is poisoned by the harness).
__global__ void zero_tail_kernel(const int* __restrict__ lens, float* __restrict__ y)
{
    int total = 0;
    #pragma unroll
    for (int i = 0; i < G_NUM; ++i) total += lens[i];
    const size_t start  = (size_t)total * OUT_DIM;
    const size_t end    = (size_t)T_TOK * OUT_DIM;
    const size_t stride = (size_t)gridDim.x * blockDim.x;
    for (size_t i = start + (size_t)blockIdx.x * blockDim.x + threadIdx.x; i < end; i += stride)
        y[i] = 0.0f;
}

extern "C" void kernel_launch(void* const* d_in, const int* in_sizes, int n_in,
                              void* d_out, int out_size, void* d_ws, size_t ws_size,
                              hipStream_t stream)
{
    (void)in_sizes; (void)n_in; (void)out_size;
    const float* x    = (const float*)d_in[0];
    const float* w    = (const float*)d_in[1];
    const float* bias = (const float*)d_in[2];
    const int*   lens = (const int*)d_in[3];
    float* y = (float*)d_out;

    zero_tail_kernel<<<2048, 256, 0, stream>>>(lens, y);

    const size_t nx = (size_t)T_TOK * IN_DIM;
    const size_t nw = (size_t)G_NUM * OUT_DIM * IN_DIM;
    const size_t need = (nx + nw) * sizeof(__bf16);   // ~100 MB

    dim3 grid(OUT_DIM / BN, CAP / BM, G_NUM);         // (32, 32, 8)

    if (ws_size >= need) {
        __bf16* xb = (__bf16*)d_ws;
        __bf16* wb = xb + nx;
        cvt_f32_to_bf16_kernel<<<4096, 256, 0, stream>>>(x, xb, nx);
        cvt_f32_to_bf16_kernel<<<8192, 256, 0, stream>>>(w, wb, nw);
        const size_t shmem = (size_t)4 * TILE_E * sizeof(__bf16); // 2 bufs x (A+B) = 72 KB
        grouped_gemm_bf16<<<grid, 256, shmem, stream>>>(xb, wb, bias, lens, y);
    } else {
        grouped_gemm_f32src<<<grid, 256, 0, stream>>>(x, w, bias, lens, y);
    }
}
